// VQ_VAE_4432406249690
// MI455X (gfx1250) — compile-verified, matrix operand order checked by reading, not run on hardware
//
#include <hip/hip_runtime.h>
#include <math.h>

// ---------------------------------------------------------------------------
// CDNA5 (gfx1250) VQ-VAE forward. Every conv / convT / VQ-cross-term is an
// implicit GEMM on v_wmma_f32_16x16x32_f16 (wave32). Inter-layer activations
// are stored f16 in workspace (net is HBM-bound: ~60 GF vs ~300 MB traffic,
// so halving activation bytes is the dominant win); weights/params arrive
// fp32 and are converted in the B-tile stage; accumulation f32.
// Block tile 128x64, BK=32, 8 waves; each wave owns a 32x32 patch:
// 2 A-frags x 2 B-frags -> 4 independent v_wmma per K-step.
// All dims are powers of two -> shift/mask addressing; taps via tables.
// ---------------------------------------------------------------------------

typedef __attribute__((ext_vector_type(16))) _Float16 v16h;
typedef __attribute__((ext_vector_type(8)))  float    v8f;

#define TILE_M 128
#define TILE_N 64
#define TILE_K 32

struct ConvParams {
    const void*  in;        // [N, IC, IH, IW], dtype TI
    const float* w;         // fp32 weights; addr = ic*icCoef + oc*ocCoef + tapOff[t]
    const float* bias;      // [OC] or null
    const float* gamma;     // BN gamma [OC] or null
    const float* beta;      // BN beta  [OC]
    const void*  residual;  // dtype TI, same shape as out, or null
    void*        out;       // [N, OC, outH, outW], dtype TO
    int IH, IW, OC;
    int stride, padY, padX;
    int icCoef, ocCoef;
    int icShift, icMask;        // IC = 1<<icShift
    int ihwShift, iwShift;      // IH*IW, IW shifts
    int owShift, owMask;        // GEMM-row grid OW
    int ohShift, ohMask;        // GEMM-row grid OH
    int ocShift;                // log2(OC)
    int outHShift, outWShift;   // full output H/W shifts
    int outYShift, outYadd;     // parity placement (convT)
    int outXShift, outXadd;
    int relu;
    int Ktotal, Mtotal;
    int tapKy[16], tapKx[16], tapOff[16];
};

template <typename TI, typename TO>
__global__ void __launch_bounds__(256)
conv_wmma_kernel(ConvParams p) {
    __shared__ _Float16 As[TILE_M][TILE_K + 8];   // [M][K]
    __shared__ _Float16 Bs[TILE_K][TILE_N + 8];   // [K][N]

    const TI* inp = (const TI*)p.in;
    const TI* res = (const TI*)p.residual;
    TO*       out = (TO*)p.out;

    const int tid  = threadIdx.x;
    const int wave = tid >> 5;
    const int lane = tid & 31;
    const int wm   = wave >> 1;   // 0..3 : 32-row patch
    const int wn   = wave & 1;    // 0..1 : 32-col patch
    const int mBase = blockIdx.x * TILE_M;
    const int nBase = blockIdx.y * TILE_N;

    // ---- A-loader invariants: thread owns col aC, rows wave+8i (i=0..15) ----
    const int aC = tid & 31;
    unsigned aOkMask = 0;
    int aYX[16];    // packed: low16 = oy*stride-padY, high16 = ox*stride-padX
    int aBase[16];  // n*IC*IH*IW
    #pragma unroll
    for (int i = 0; i < 16; ++i) {
        int m  = mBase + wave + 8 * i;
        int ok = m < p.Mtotal;
        int mm = ok ? m : 0;
        int ox = mm & p.owMask;
        int t2 = mm >> p.owShift;
        int oy = t2 & p.ohMask;
        int n  = t2 >> p.ohShift;
        int yb = oy * p.stride - p.padY;
        int xb = ox * p.stride - p.padX;
        aYX[i]   = (yb & 0xffff) | (xb << 16);
        aBase[i] = (n << p.icShift) << p.ihwShift;
        aOkMask |= (unsigned)ok << i;
    }
    // ---- B-loader invariants: thread owns col bN, rows (tid>>6)+4i ----
    const int bN      = tid & 63;
    const int bKT     = tid >> 6;                 // 0..3
    const int bOc     = nBase + bN;
    const int bOcOk   = bOc < p.OC;
    const int bOcTerm = bOc * p.ocCoef;

    v8f acc00 = {0.f,0.f,0.f,0.f,0.f,0.f,0.f,0.f};
    v8f acc01 = acc00, acc10 = acc00, acc11 = acc00;

    const int aRow0 = wm * 32 + (lane & 15);
    const int aRow1 = aRow0 + 16;
    const int kh    = (lane >> 4) << 3;           // 0 or 8
    const int bCol  = wn * 32 + (lane & 15);

    for (int k0 = 0; k0 < p.Ktotal; k0 += TILE_K) {
        // ---- A tile: 128x32 gathered TI -> f16 (one k-column per thread)
        {
            const int k   = k0 + aC;
            const int kOk = k < p.Ktotal;
            int ky = 0, kx = 0, icTerm = 0;
            if (kOk) {
                int ic = k & p.icMask;
                int tp = k >> p.icShift;
                ky = p.tapKy[tp];
                kx = p.tapKx[tp];
                icTerm = ic << p.ihwShift;
            }
            #pragma unroll
            for (int i = 0; i < 16; ++i) {
                _Float16 hv = (_Float16)0.f;
                if (kOk && ((aOkMask >> i) & 1u)) {
                    int iy = (int)(short)aYX[i] + ky;
                    int ix = (aYX[i] >> 16) + kx;
                    if ((unsigned)iy < (unsigned)p.IH &&
                        (unsigned)ix < (unsigned)p.IW)
                        hv = (_Float16)inp[aBase[i] + icTerm +
                                           (iy << p.iwShift) + ix];
                }
                As[wave + 8 * i][aC] = hv;
            }
        }
        // ---- B tile: 32x64 fp32 weights -> f16 (one n-column per thread)
        #pragma unroll
        for (int i = 0; i < 8; ++i) {
            int kk = bKT + 4 * i;
            int k  = k0 + kk;
            float v = 0.f;
            if (bOcOk && k < p.Ktotal) {
                int ic = k & p.icMask;
                int tp = k >> p.icShift;
                v = p.w[ic * p.icCoef + bOcTerm + p.tapOff[tp]];
            }
            Bs[kk][bN] = (_Float16)v;
        }
        __syncthreads();

        // ---- fragments per ISA 16-bit layouts, 4 WMMA per wave ----
        v16h a0, a1, b0, b1;
        #pragma unroll
        for (int i = 0; i < 8; ++i) {
            a0[i]     = As[aRow0][kh + i];
            a0[8 + i] = As[aRow0][16 + kh + i];
            a1[i]     = As[aRow1][kh + i];
            a1[8 + i] = As[aRow1][16 + kh + i];
        }
        #pragma unroll
        for (int i = 0; i < 8; ++i) {
            b0[i]     = Bs[kh + i][bCol];
            b0[8 + i] = Bs[16 + kh + i][bCol];
            b1[i]     = Bs[kh + i][bCol + 16];
            b1[8 + i] = Bs[16 + kh + i][bCol + 16];
        }
        acc00 = __builtin_amdgcn_wmma_f32_16x16x32_f16(
                    false, a0, false, b0, (short)0, acc00, false, false);
        acc01 = __builtin_amdgcn_wmma_f32_16x16x32_f16(
                    false, a0, false, b1, (short)0, acc01, false, false);
        acc10 = __builtin_amdgcn_wmma_f32_16x16x32_f16(
                    false, a1, false, b0, (short)0, acc10, false, false);
        acc11 = __builtin_amdgcn_wmma_f32_16x16x32_f16(
                    false, a1, false, b1, (short)0, acc11, false, false);
        __syncthreads();
    }

    // ---- epilogue: bias + BN(eval) + ReLU + residual (f32), store TO ----
    const float kBNInv = 0.9999950000374997f;    // rsqrt(1 + 1e-5)
    auto epi = [&](v8f acc, int mf, int nf) {
        const int oc = nBase + wn * 32 + nf * 16 + (lane & 15);
        if (oc >= p.OC) return;
        const float bv = p.bias  ? p.bias[oc] : 0.f;
        const float g  = p.gamma ? p.gamma[oc] * kBNInv : 1.f;
        const float bt = p.gamma ? p.beta[oc] : 0.f;
        #pragma unroll
        for (int r = 0; r < 8; ++r) {
            int m = mBase + wm * 32 + mf * 16 + ((lane >> 4) << 3) + r;
            if (m < p.Mtotal) {
                float v = acc[r] + bv;
                v = v * g + bt;
                if (p.relu) v = fmaxf(v, 0.f);
                int ox = m & p.owMask;
                int t2 = m >> p.owShift;
                int oy = t2 & p.ohMask;
                int n  = t2 >> p.ohShift;
                int oyo = (oy << p.outYShift) + p.outYadd;
                int oxo = (ox << p.outXShift) + p.outXadd;
                int chan = (n << p.ocShift) + oc;
                int oidx = (((chan << p.outHShift) + oyo) << p.outWShift) + oxo;
                if (res) v += (float)res[oidx];
                out[oidx] = (TO)v;
            }
        }
    };
    epi(acc00, 0, 0);
    epi(acc01, 0, 1);
    epi(acc10, 1, 0);
    epi(acc11, 1, 1);
}

// ---------------------------------------------------------------------------
// Small helper kernels
// ---------------------------------------------------------------------------
__global__ void zero_kernel(float* p, int n) {
    int i = blockIdx.x * blockDim.x + threadIdx.x;
    if (i < n) p[i] = 0.f;
}

__global__ void relu_h_kernel(const _Float16* in, _Float16* out, int n) {
    for (int i = blockIdx.x * blockDim.x + threadIdx.x; i < n;
         i += gridDim.x * blockDim.x) {
        float v = (float)in[i];
        out[i] = (_Float16)fmaxf(v, 0.f);
    }
}

__global__ void codebook_sq_kernel(const float* cb, float* w2) {
    int k = threadIdx.x;  // 128 threads
    float s = 0.f;
    for (int d = 0; d < 128; ++d) { float v = cb[k * 128 + d]; s += v * v; }
    w2[k] = s;
}

// per encoder pixel: argmin_k (||w_k||^2 - 2*cross), gather code row,
// straight-through forward value (== quantized), loss partials + histogram.
__global__ void vq_assign_kernel(const _Float16* z, const float* cross,
                                 const float* cb, const float* w2,
                                 _Float16* zq, float* vq_sum, float* hist) {
    int p = blockIdx.x * 256 + threadIdx.x;      // 32*16*16 = 8192 pixels
    if (p >= 32 * 256) return;
    int hw = p & 255;
    int b  = p >> 8;
    int best = 0;
    float bestScore = 3.4e38f;
    for (int k = 0; k < 128; ++k) {
        float s = w2[k] - 2.f * cross[(b * 128 + k) * 256 + hw];
        if (s < bestScore) { bestScore = s; best = k; }
    }
    float sum = 0.f;
    for (int d = 0; d < 128; ++d) {
        float q = cb[best * 128 + d];
        int zi = (b * 128 + d) * 256 + hw;
        float diff = q - (float)z[zi];
        sum += diff * diff;
        zq[zi] = (_Float16)q;
    }
    atomicAdd(vq_sum, sum);
    atomicAdd(&hist[best], 1.f);
}

__global__ void sqerr_reduce_kernel(const float* a, const float* b,
                                    float* out, int n) {
    __shared__ float sm[256];
    float s = 0.f;
    for (int i = blockIdx.x * 256 + threadIdx.x; i < n; i += gridDim.x * 256) {
        float d = a[i] - b[i];
        s += d * d;
    }
    sm[threadIdx.x] = s;
    __syncthreads();
    for (int off = 128; off > 0; off >>= 1) {
        if (threadIdx.x < off) sm[threadIdx.x] += sm[threadIdx.x + off];
        __syncthreads();
    }
    if (threadIdx.x == 0) atomicAdd(out, sm[0]);
}

__global__ void finalize_kernel(const float* S, float* out4) {
    if (threadIdx.x != 0 || blockIdx.x != 0) return;
    float recon = S[1] / 1048576.f;
    float closs = 1.25f * S[0] / 1048576.f;
    float ent = 0.f;
    for (int k = 0; k < 128; ++k) {
        float pk = S[2 + k] / 8192.f;
        ent += pk * logf(pk + 1e-10f);
    }
    out4[0] = recon;
    out4[1] = closs;
    out4[2] = recon + closs;
    out4[3] = expf(-ent);
}

// ---------------------------------------------------------------------------
// Host-side launch helpers
// ---------------------------------------------------------------------------
static int ilog2i(int x) { int s = 0; while ((1 << s) < x) ++s; return s; }

template <typename TI, typename TO>
static void launchConv(hipStream_t s, const void* in, const float* w,
                       const float* bias, const float* gamma, const float* beta,
                       const void* residual, void* out,
                       int N, int IC, int IH, int IW, int OC,
                       int KH, int KW, int stride, int pad, int relu) {
    ConvParams p{};
    p.in = in; p.w = w; p.bias = bias; p.gamma = gamma; p.beta = beta;
    p.residual = residual; p.out = out;
    int OH = (IH + 2 * pad - KH) / stride + 1;
    int OW = (IW + 2 * pad - KW) / stride + 1;
    p.IH = IH; p.IW = IW; p.OC = OC;
    p.stride = stride; p.padY = pad; p.padX = pad;
    p.icCoef = KH * KW; p.ocCoef = IC * KH * KW;
    p.icShift = ilog2i(IC); p.icMask = IC - 1;
    p.ihwShift = ilog2i(IH) + ilog2i(IW); p.iwShift = ilog2i(IW);
    p.owShift = ilog2i(OW); p.owMask = OW - 1;
    p.ohShift = ilog2i(OH); p.ohMask = OH - 1;
    p.ocShift = ilog2i(OC);
    p.outHShift = ilog2i(OH); p.outWShift = ilog2i(OW);
    p.outYShift = 0; p.outYadd = 0; p.outXShift = 0; p.outXadd = 0;
    p.relu = relu; p.Ktotal = IC * KH * KW; p.Mtotal = N * OH * OW;
    for (int t = 0; t < KH * KW; ++t) {
        p.tapKy[t] = t / KW; p.tapKx[t] = t % KW; p.tapOff[t] = t;
    }
    dim3 grid((p.Mtotal + TILE_M - 1) / TILE_M, (OC + TILE_N - 1) / TILE_N);
    conv_wmma_kernel<TI, TO><<<grid, 256, 0, s>>>(p);
}

// ConvTranspose2d(k=4, s=2, p=1), weight (IC, OC, 4, 4), as 4 parity-class
// standard 2x2-tap convs on the same WMMA kernel. f16 -> f16.
static void launchConvT(hipStream_t s, const void* in, const float* w,
                        const float* bias, void* out,
                        int N, int IC, int IH, int IW, int OC, int relu) {
    const int OHf = IH * 2, OWf = IW * 2;
    for (int py = 0; py < 2; ++py)
        for (int px = 0; px < 2; ++px) {
            ConvParams p{};
            p.in = in; p.w = w; p.bias = bias; p.out = out;
            p.IH = IH; p.IW = IW; p.OC = OC;
            p.stride = 1; p.padY = 1 - py; p.padX = 1 - px;
            p.icCoef = OC * 16;           // weight (IC, OC, 4, 4)
            p.ocCoef = 16;
            p.icShift = ilog2i(IC); p.icMask = IC - 1;
            p.ihwShift = ilog2i(IH) + ilog2i(IW); p.iwShift = ilog2i(IW);
            p.owShift = ilog2i(IW); p.owMask = IW - 1;   // parity grid = IHxIW
            p.ohShift = ilog2i(IH); p.ohMask = IH - 1;
            p.ocShift = ilog2i(OC);
            p.outHShift = ilog2i(OHf); p.outWShift = ilog2i(OWf);
            p.outYShift = 1; p.outYadd = py;
            p.outXShift = 1; p.outXadd = px;
            p.relu = relu; p.Ktotal = IC * 4; p.Mtotal = N * IH * IW;
            for (int t = 0; t < 2; ++t)
                for (int u = 0; u < 2; ++u) {
                    p.tapKy[t * 2 + u]  = t;
                    p.tapKx[t * 2 + u]  = u;
                    p.tapOff[t * 2 + u] = (3 - py - 2 * t) * 4 + (3 - px - 2 * u);
                }
            dim3 grid((p.Mtotal + TILE_M - 1) / TILE_M,
                      (OC + TILE_N - 1) / TILE_N);
            conv_wmma_kernel<_Float16, _Float16><<<grid, 256, 0, s>>>(p);
        }
}

// ---------------------------------------------------------------------------
// Entry point
// ---------------------------------------------------------------------------
extern "C" void kernel_launch(void* const* d_in, const int* in_sizes, int n_in,
                              void* d_out, int out_size, void* d_ws,
                              size_t ws_size, hipStream_t stream) {
    (void)in_sizes; (void)n_in; (void)out_size; (void)ws_size;

    const float* P[44];
    for (int i = 0; i < 44; ++i) P[i] = (const float*)d_in[i];
    const float* inputs = P[0];
    const float *e0w = P[1],  *e0b = P[2];
    const float *e1w = P[3],  *e1b = P[4],  *g1 = P[5],  *b1 = P[6];
    const float *e2w = P[7],  *e2b = P[8],  *g2 = P[9],  *b2 = P[10];
    const float *e3w = P[11], *e3b = P[12], *g3 = P[13], *b3 = P[14];
    const float *e4w = P[15], *e4b = P[16], *g4 = P[17], *b4 = P[18];
    const float *codebook = P[19];
    const float *d0w = P[20], *d0b = P[21];
    const float *d1w = P[22], *d1b = P[23];
    const float *d2w = P[24], *d2b = P[25];
    const float *d3w = P[26], *d3b = P[27];
    const float *raw[2]  = {P[28], P[36]}, *rab[2]  = {P[29], P[37]};
    const float *rag[2]  = {P[30], P[38]}, *rabt[2] = {P[31], P[39]};
    const float *rbw[2]  = {P[32], P[40]}, *rbb[2]  = {P[33], P[41]};
    const float *rbg[2]  = {P[34], P[42]}, *rbbt[2] = {P[35], P[43]};

    // f16 activation workspace (byte offsets; ~99 MB total)
    char* wsb = (char*)d_ws;
    _Float16* A0h = (_Float16*)(wsb);               // 67,108,864 B : e0 out
    _Float16* A1h = (_Float16*)(wsb + 67108864u);   // 16,777,216 B : e1 out
    _Float16* A2h = (_Float16*)(wsb + 83886080u);   //  8,388,608 B : e2 out
    _Float16* A3h = (_Float16*)(wsb + 92274688u);   //  2,097,152 B : e3 out
    _Float16* X0h = (_Float16*)(wsb + 94371840u);   //  2,097,152 B : res ping
    _Float16* X1h = (_Float16*)(wsb + 96468992u);   //  2,097,152 B : res pong / zq
    float*    S   = (float*)  (wsb + 98566144u);    //  258 floats of scalars
    _Float16* H512h = A2h;                          // reuse: 512ch hidden (8 MB)
    float*    crossF = (float*)A2h;                 // reuse: VQ cross f32 (4 MB)
    _Float16* H128h = A3h;                          // reuse: relu(x)
    _Float16* D0h = A1h;                            // reuse: d0 out (4 MB)
    _Float16* D1h = A0h;                            // reuse: d1 out (8 MB)
    _Float16* D2h = (_Float16*)(wsb + 8388608u);    // reuse: d2 out (32 MB)
    float* dec = (float*)d_out;                     // decoded [32,2,128,128]

    zero_kernel<<<1, 256, 0, stream>>>(S, 130);

    // ---- encoder ----
    launchConv<float, _Float16>(stream, inputs, e0w, e0b, nullptr, nullptr,
               nullptr, A0h, 32, 2, 128, 128, 64, 1, 1, 1, 0, 0);
    launchConv<_Float16, _Float16>(stream, A0h, e1w, e1b, g1, b1,
               nullptr, A1h, 32, 64, 128, 128, 64, 4, 4, 2, 1, 1);
    launchConv<_Float16, _Float16>(stream, A1h, e2w, e2b, g2, b2,
               nullptr, A2h, 32, 64, 64, 64, 128, 4, 4, 2, 1, 1);
    launchConv<_Float16, _Float16>(stream, A2h, e3w, e3b, g3, b3,
               nullptr, A3h, 32, 128, 32, 32, 128, 4, 4, 2, 1, 1);
    launchConv<_Float16, _Float16>(stream, A3h, e4w, e4b, g4, b4,
               nullptr, X0h, 32, 128, 16, 16, 128, 3, 3, 1, 1, 0);

    // ---- residual stack ----
    _Float16* xin = X0h; _Float16* xout = X1h;
    for (int i = 0; i < 2; ++i) {
        relu_h_kernel<<<1024, 256, 0, stream>>>(xin, H128h, 1048576);
        launchConv<_Float16, _Float16>(stream, H128h, raw[i], rab[i], rag[i],
                   rabt[i], nullptr, H512h, 32, 128, 16, 16, 512, 3, 3, 1, 1, 1);
        launchConv<_Float16, _Float16>(stream, H512h, rbw[i], rbb[i], rbg[i],
                   rbbt[i], xin, xout, 32, 512, 16, 16, 128, 1, 1, 1, 0, 0);
        _Float16* t = xin; xin = xout; xout = t;
    }
    _Float16* z  = xin;
    _Float16* zq = xout;

    // ---- vector quantization ----
    // cross[b,k,h,w] = sum_d z[b,d,h,w]*cb[k,d]  (1x1 conv, f16 in, f32 out)
    launchConv<_Float16, float>(stream, z, codebook, nullptr, nullptr, nullptr,
               nullptr, crossF, 32, 128, 16, 16, 128, 1, 1, 1, 0, 0);
    codebook_sq_kernel<<<1, 128, 0, stream>>>(codebook, S + 130);
    vq_assign_kernel<<<32, 256, 0, stream>>>(z, crossF, codebook, S + 130,
                                             zq, S + 0, S + 2);

    // ---- decoder ----
    launchConvT(stream, zq,  d0w, d0b, D0h, 32, 128, 16, 16, 64, 1);
    launchConvT(stream, D0h, d1w, d1b, D1h, 32, 64, 32, 32, 32, 1);
    launchConvT(stream, D1h, d2w, d2b, D2h, 32, 32, 64, 64, 32, 1);
    launchConv<_Float16, float>(stream, D2h, d3w, d3b, nullptr, nullptr,
               nullptr, dec, 32, 32, 128, 128, 2, 1, 1, 1, 0, 0);

    // ---- losses ----
    sqerr_reduce_kernel<<<1024, 256, 0, stream>>>(dec, inputs, S + 1, 1048576);
    finalize_kernel<<<1, 32, 0, stream>>>(S, dec + 1048576);
}
